// FlashARModelBase_9646496547210
// MI455X (gfx1250) — matrix operand order repaired and checked by reading, not compiled, not run on hardware
//
#include <hip/hip_runtime.h>
#include <math.h>

// ---------------------------------------------------------------------------
// Types for CDNA5 WMMA (wave32): v_wmma_f32_16x16x32_bf16
// ---------------------------------------------------------------------------
typedef __attribute__((ext_vector_type(16))) __bf16 v16bf;
typedef __attribute__((ext_vector_type(8)))  float  v8f;

union Frag16 { unsigned u[8]; v16bf v; };
union V4U { uint4 q; unsigned short s[8]; };

__device__ inline v8f vzero8() {
  v8f z;
#pragma unroll
  for (int i = 0; i < 8; ++i) z[i] = 0.0f;
  return z;
}

__device__ inline unsigned short f2bf(float x) {
  union { float f; unsigned u; } c; c.f = x;
  unsigned r = c.u + 0x7FFFu + ((c.u >> 16) & 1u);   // round-nearest-even
  return (unsigned short)(r >> 16);
}

__device__ inline float gelu_tanh(float x) {
  float x3 = x * x * x;
  return 0.5f * x * (1.0f + tanhf(0.7978845608028654f * (x + 0.044715f * x3)));
}

__device__ inline v8f wmma_bf16(v16bf a, v16bf b, v8f c) {
  // (neg_a, A, neg_b, B, c_mod, C, reuse_a, reuse_b)
  return __builtin_amdgcn_wmma_f32_16x16x32_bf16(false, a, false, b, (short)0, c,
                                                 false, false);
}

// ---------------------------------------------------------------------------
// CDNA5 async global->LDS copy (ASYNCcnt-tracked, no VGPR round trip).
// Per the ISA doc,
// the immediate offset advances BOTH addresses, so one (vdst, vaddr) pair can
// issue several 16B chunks with different offset: immediates (no VALU adds).
// Generic shared pointer truncated to 32 bits is the LDS offset.
// ---------------------------------------------------------------------------
template <int BYTE_OFF>
__device__ inline void async_ld_b128(unsigned short* lds_dst,
                                     const unsigned short* gsrc) {
  unsigned loff = (unsigned)(unsigned long long)lds_dst;
  unsigned long long ga = (unsigned long long)gsrc;
  asm volatile("global_load_async_to_lds_b128 %0, %1, off offset:%2"
               :
               : "v"(loff), "v"(ga), "n"(BYTE_OFF)
               : "memory");
}
__device__ inline void async_wait0() {
  asm volatile("s_wait_asynccnt 0x0" ::: "memory");
}

// A 16x32 bf16 fragment in this code's contiguous k order: lanes 0-15 hold K=0..15, lanes 16-31 K=16..31
__device__ inline v16bf load_frag_a(const unsigned short* base, int row0, int stride,
                                    int kofs) {
  int lane = threadIdx.x & 31;
  int r = row0 + (lane & 15);
  int half = lane >> 4;
  const unsigned short* p = base + r * stride + kofs;
  Frag16 f;
#pragma unroll
  for (int i = 0; i < 8; ++i) {
    int k = half * 16 + i * 2;  // half*16+i*2
    f.u[i] = *(const unsigned*)(p + k);
  }
  return f.v;
}

// B fragment 32x16 bf16 from B^T rows (row index = N), same k order as A: lanes 0-15 K=0..15,
// lanes 16-31 K=16..31, two K per VGPR.
__device__ inline v16bf load_frag_b(const unsigned short* base, int row0, int stride,
                                    int kofs) {
  int lane = threadIdx.x & 31;
  int r = row0 + (lane & 15);
  int half = lane >> 4;
  const unsigned short* p = base + r * stride + kofs + half * 16;
  Frag16 f;
#pragma unroll
  for (int i = 0; i < 8; ++i) f.u[i] = *(const unsigned*)(p + i * 2);
  return f.v;
}

// ---------------------------------------------------------------------------
// Embedding: h[b,t,:] = concat(tok_emb[x_shifted], pos_emb[t])
// ---------------------------------------------------------------------------
__global__ void embed_kernel(const int* __restrict__ x, const float* __restrict__ tok,
                             const float* __restrict__ pos, float* __restrict__ h,
                             int B, int T, int Dm, int TD) {
  int total = B * T * Dm;
  for (int idx = blockIdx.x * blockDim.x + threadIdx.x; idx < total;
       idx += gridDim.x * blockDim.x) {
    int d = idx % Dm;
    int t = (idx / Dm) % T;
    int b = idx / (Dm * T);
    int tokid = (t == 0) ? 1 : x[b * T + t - 1];
    float v = (d < TD) ? tok[tokid * TD + d]
                       : pos[(size_t)t * (Dm - TD) + (d - TD)];
    h[idx] = v;
  }
}

// ---------------------------------------------------------------------------
// f32 -> bf16 weight conversion
// ---------------------------------------------------------------------------
__global__ void f32_to_bf16_kernel(const float* __restrict__ in,
                                   unsigned short* __restrict__ out, int n) {
  int i = blockIdx.x * blockDim.x + threadIdx.x;
  if (i < n) out[i] = f2bf(in[i]);
}

// ---------------------------------------------------------------------------
// LayerNorm: one block per row; bf16 out, optional f32 out (in-place allowed)
// ---------------------------------------------------------------------------
__global__ __launch_bounds__(256) void ln_kernel(const float* __restrict__ h,
                                                 const float* __restrict__ g,
                                                 const float* __restrict__ beta,
                                                 unsigned short* __restrict__ outb,
                                                 float* __restrict__ outf, int Dm) {
  __shared__ float red[256];
  int row = blockIdx.x, tid = threadIdx.x;
  const float* xr = h + (size_t)row * Dm;
  float s = 0.f;
  for (int i = tid; i < Dm; i += 256) s += xr[i];
  red[tid] = s; __syncthreads();
  for (int off = 128; off > 0; off >>= 1) {
    if (tid < off) red[tid] += red[tid + off];
    __syncthreads();
  }
  float mu = red[0] / Dm; __syncthreads();
  float s2 = 0.f;
  for (int i = tid; i < Dm; i += 256) { float d = xr[i] - mu; s2 += d * d; }
  red[tid] = s2; __syncthreads();
  for (int off = 128; off > 0; off >>= 1) {
    if (tid < off) red[tid] += red[tid + off];
    __syncthreads();
  }
  float rstd = rsqrtf(red[0] / Dm + 1e-5f);
  for (int i = tid; i < Dm; i += 256) {
    float y = (xr[i] - mu) * rstd * g[i] + beta[i];
    outb[(size_t)row * Dm + i] = f2bf(y);
    if (outf) outf[(size_t)row * Dm + i] = y;
  }
}

// ---------------------------------------------------------------------------
// WMMA GEMM: C[M,N] = A[M,K] * W[N,K]^T (+bias, epilogue)
// Block tile 128x64, 8 waves, each wave 32x32 (2x2 WMMA frags), K-step 64
// (8 WMMAs per barrier pair). Tiles staged via async global->LDS.
// ep: 0 = QKV (scale cols < qcols by 0.125, bf16 out)
//     1 = GELU (bf16 out)
//     2 = residual add into resid (f32)
// ---------------------------------------------------------------------------
#define GBM 128
#define GBN 64
#define GBK 64
__global__ __launch_bounds__(256) void gemm_bf16_wmma(
    const unsigned short* __restrict__ A, const unsigned short* __restrict__ W,
    const float* __restrict__ bias, float* __restrict__ resid,
    unsigned short* __restrict__ outb, int M, int N, int K, int ep, int qcols) {
  __shared__ unsigned short sA[GBM * GBK];   // 16 KB
  __shared__ unsigned short sB[GBN * GBK];   // 8 KB
  int tid = threadIdx.x;
  int wid = tid >> 5, lane = tid & 31;
  int bm = blockIdx.x * GBM, bn = blockIdx.y * GBN;
  int wm = (wid & 3) * 32;   // wave row offset in tile
  int wn = (wid >> 2) * 32;  // wave col offset in tile
  v8f c00 = vzero8(), c01 = vzero8(), c10 = vzero8(), c11 = vzero8();

  int ra = tid >> 1, ca = (tid & 1) * 32;    // A: 32 elems/thread
  int rb = tid >> 2, cb = (tid & 3) * 16;    // W: 16 elems/thread
  const unsigned short* pA = A + (size_t)(bm + ra) * K + ca;
  const unsigned short* pW = W + (size_t)(bn + rb) * K + cb;
  unsigned short* lA = sA + ra * GBK + ca;
  unsigned short* lW = sB + rb * GBK + cb;

  for (int k0 = 0; k0 < K; k0 += GBK) {
    // async stage tiles into LDS (imm offset advances both LDS+global addr)
    async_ld_b128<0>(lA, pA);
    async_ld_b128<16>(lA, pA);
    async_ld_b128<32>(lA, pA);
    async_ld_b128<48>(lA, pA);
    async_ld_b128<0>(lW, pW);
    async_ld_b128<16>(lW, pW);
    if (k0 + GBK < K) {  // prefetch next K-tile (global_prefetch_b8)
      __builtin_prefetch(pA + GBK, 0, 0);
      __builtin_prefetch(pW + GBK, 0, 0);
    }
    async_wait0();
    __syncthreads();
#pragma unroll
    for (int kk = 0; kk < GBK; kk += 32) {
      v16bf a0 = load_frag_a(sA, wm, GBK, kk);
      v16bf a1 = load_frag_a(sA, wm + 16, GBK, kk);
      v16bf b0 = load_frag_b(sB, wn, GBK, kk);
      v16bf b1 = load_frag_b(sB, wn + 16, GBK, kk);
      c00 = wmma_bf16(a0, b0, c00);
      c01 = wmma_bf16(a0, b1, c01);
      c10 = wmma_bf16(a1, b0, c10);
      c11 = wmma_bf16(a1, b1, c11);
    }
    __syncthreads();
    pA += GBK;
    pW += GBK;
  }

  int half = lane >> 4, ln = lane & 15;
#pragma unroll
  for (int s = 0; s < 2; ++s) {
    v8f* crow0 = (s == 0) ? &c00 : &c10;
    v8f* crow1 = (s == 0) ? &c01 : &c11;
#pragma unroll
    for (int t = 0; t < 2; ++t) {
      v8f cc = (t == 0) ? *crow0 : *crow1;
      int n = bn + wn + t * 16 + ln;
      float bi = bias ? bias[n] : 0.0f;
#pragma unroll
      for (int v = 0; v < 8; ++v) {
        int m = bm + wm + s * 16 + v + half * 8;
        float val = cc[v] + bi;
        size_t o = (size_t)m * N + n;
        if (ep == 0) {
          if (n < qcols) val *= 0.125f;  // q pre-scaled by 1/sqrt(64)
          outb[o] = f2bf(val);
        } else if (ep == 1) {
          outb[o] = f2bf(gelu_tanh(val));
        } else {
          resid[o] += val;
        }
      }
    }
  }
}

// ---------------------------------------------------------------------------
// Flash attention: block = (query-tile of 64, head, batch); 8 waves.
// qkv: [B*T, 3*D] bf16 (q pre-scaled). Adds O into h.
// ---------------------------------------------------------------------------
__global__ __launch_bounds__(256) void flash_attn_wmma(
    const unsigned short* __restrict__ qkv, float* __restrict__ h, int T, int D) {
  const int HD = 64;
  __shared__ unsigned short sQ[64 * 64];
  __shared__ unsigned short sK[64 * 64];
  __shared__ unsigned short sVT[64 * 64];  // [d][n]
  __shared__ float sS[64 * 64];
  __shared__ unsigned short sP[64 * 64];
  __shared__ float sM[64], sL[64], sAl[64];

  int tid = threadIdx.x, wid = tid >> 5, lane = tid & 31;
  int mt = blockIdx.x, hd = blockIdx.y, b = blockIdx.z;
  int m0 = mt * 64;
  size_t rs = (size_t)3 * D;
  const unsigned short* qb = qkv + (size_t)b * T * rs + hd * HD;
  const unsigned short* kb = qb + D;
  const unsigned short* vb = qb + 2 * D;

  int rt = tid >> 2, ct = (tid & 3) * 16;  // 16 elems/thread for 64x64 tiles
  {  // Q tile via async global->LDS
    const unsigned short* src = qb + (size_t)(m0 + rt) * rs + ct;
    unsigned short* dst = sQ + rt * 64 + ct;
    async_ld_b128<0>(dst, src);
    async_ld_b128<16>(dst, src);
  }
  if (tid < 64) { sM[tid] = -3.0e38f; sL[tid] = 0.0f; }
  async_wait0();
  __syncthreads();

  int wm = (wid & 3) * 16;    // S/O row offset
  int wns = (wid >> 2) * 32;  // S col offset
  int wd = (wid >> 2) * 32;   // O d-col offset
  int half = lane >> 4, ln = lane & 15;
  v8f o0 = vzero8(), o1 = vzero8();

  for (int j = 0; j <= mt; ++j) {
    {  // K tile via async global->LDS
      const unsigned short* src = kb + (size_t)(j * 64 + rt) * rs + ct;
      unsigned short* dst = sK + rt * 64 + ct;
      async_ld_b128<0>(dst, src);
      async_ld_b128<16>(dst, src);
    }
#pragma unroll
    for (int it = 0; it < 2; ++it) {  // V tile, transposed into sVT
      int n = (tid >> 3) + it * 32, d0 = (tid & 7) * 8;
      V4U u;
      u.q = *(const uint4*)(vb + (size_t)(j * 64 + n) * rs + d0);
#pragma unroll
      for (int i = 0; i < 8; ++i) sVT[(d0 + i) * 64 + n] = u.s[i];
    }
    async_wait0();
    __syncthreads();

    // S = Q K^T (16x32 patch per wave), K-dim = head_dim = 64
    v8f s0 = vzero8(), s1 = vzero8();
#pragma unroll
    for (int d0 = 0; d0 < 64; d0 += 32) {
      v16bf aq = load_frag_a(sQ, wm, 64, d0);
      v16bf bk0 = load_frag_b(sK, wns, 64, d0);
      v16bf bk1 = load_frag_b(sK, wns + 16, 64, d0);
      s0 = wmma_bf16(aq, bk0, s0);
      s1 = wmma_bf16(aq, bk1, s1);
    }
#pragma unroll
    for (int v = 0; v < 8; ++v) {
      int r = wm + v + half * 8;
      sS[r * 64 + wns + ln] = s0[v];
      sS[r * 64 + wns + 16 + ln] = s1[v];
    }
    __syncthreads();

    // online softmax, one thread per row
    if (tid < 64) {
      int r = tid, mg = m0 + r;
      int lim = (j * 64 + 63 <= mg) ? 64 : (mg - j * 64 + 1);  // causal
      float mx = sM[r];
      for (int c = 0; c < lim; ++c) mx = fmaxf(mx, sS[r * 64 + c]);
      float al = __expf(sM[r] - mx);
      float sum = 0.f;
      for (int c = 0; c < 64; ++c) {
        float p = (c < lim) ? __expf(sS[r * 64 + c] - mx) : 0.0f;
        sP[r * 64 + c] = f2bf(p);
        sum += p;
      }
      sL[r] = sL[r] * al + sum;
      sM[r] = mx;
      sAl[r] = al;
    }
    __syncthreads();

    // rescale O accumulators, then O += P * V
#pragma unroll
    for (int v = 0; v < 8; ++v) {
      float al = sAl[wm + v + half * 8];
      o0[v] *= al;
      o1[v] *= al;
    }
#pragma unroll
    for (int n0 = 0; n0 < 64; n0 += 32) {
      v16bf ap = load_frag_a(sP, wm, 64, n0);
      v16bf bv0 = load_frag_b(sVT, wd, 64, n0);
      v16bf bv1 = load_frag_b(sVT, wd + 16, 64, n0);
      o0 = wmma_bf16(ap, bv0, o0);
      o1 = wmma_bf16(ap, bv1, o1);
    }
    __syncthreads();
  }

#pragma unroll
  for (int v = 0; v < 8; ++v) {
    int r = wm + v + half * 8;
    float inv = 1.0f / sL[r];
    size_t base = ((size_t)b * T + m0 + r) * D + hd * HD;
    h[base + wd + ln] += o0[v] * inv;        // residual add (unique owner)
    h[base + wd + 16 + ln] += o1[v] * inv;
  }
}

// ---------------------------------------------------------------------------
// Output head: logits = h @ out_w^T + out_b, then log_softmax (M_CAT = 2)
// ---------------------------------------------------------------------------
__global__ void logits_kernel(const float* __restrict__ h, const float* __restrict__ ow,
                              const float* __restrict__ ob, float* __restrict__ out,
                              int rows, int Dm) {
  int r = blockIdx.x * blockDim.x + threadIdx.x;
  if (r >= rows) return;
  const float* x = h + (size_t)r * Dm;
  float z0 = ob[0], z1 = ob[1];
  for (int i = 0; i < Dm; ++i) {
    float v = x[i];
    z0 += v * ow[i];
    z1 += v * ow[Dm + i];
  }
  float mx = fmaxf(z0, z1);
  float lse = mx + __logf(__expf(z0 - mx) + __expf(z1 - mx));
  out[2 * r] = z0 - lse;
  out[2 * r + 1] = z1 - lse;
}

// ---------------------------------------------------------------------------
// Host orchestration
// ---------------------------------------------------------------------------
extern "C" void kernel_launch(void* const* d_in, const int* in_sizes, int n_in,
                              void* d_out, int out_size, void* d_ws, size_t ws_size,
                              hipStream_t stream) {
  (void)in_sizes; (void)n_in; (void)out_size; (void)ws_size;
  const int B = 4, T = 1024, D = 1024, TD = 256, DFF = 4096, H = 16, L = 4;
  const int ROWS = B * T;

  const int*   x     = (const int*)d_in[0];
  const float* tok   = (const float*)d_in[1];
  const float* pos   = (const float*)d_in[2];
  const float* wqkv  = (const float*)d_in[3];
  const float* bqkv  = (const float*)d_in[4];
  const float* ln1g  = (const float*)d_in[5];
  const float* ln1b  = (const float*)d_in[6];
  const float* ln2g  = (const float*)d_in[7];
  const float* ln2b  = (const float*)d_in[8];
  const float* ffw1  = (const float*)d_in[9];
  const float* ffb1  = (const float*)d_in[10];
  const float* ffw2  = (const float*)d_in[11];
  const float* ffb2  = (const float*)d_in[12];
  const float* outw  = (const float*)d_in[13];
  const float* outbb = (const float*)d_in[14];

  char* ws = (char*)d_ws;
  size_t off = 0;
  auto alloc = [&](size_t bytes) -> void* {
    void* p = ws + off;
    off = (off + bytes + 255) & ~(size_t)255;
    return p;
  };
  float*          h    = (float*)alloc((size_t)ROWS * D * 4);
  unsigned short* act  = (unsigned short*)alloc((size_t)ROWS * D * 2);
  unsigned short* qkvb = (unsigned short*)alloc((size_t)ROWS * 3 * D * 2);
  unsigned short* gel  = (unsigned short*)alloc((size_t)ROWS * DFF * 2);
  unsigned short* wqb  = (unsigned short*)alloc((size_t)3 * D * D * 2);
  unsigned short* w1b  = (unsigned short*)alloc((size_t)DFF * D * 2);
  unsigned short* w2b  = (unsigned short*)alloc((size_t)D * DFF * 2);

  embed_kernel<<<4096, 256, 0, stream>>>(x, tok, pos, h, B, T, D, TD);

  for (int l = 0; l < L; ++l) {
    int nq = 3 * D * D, n1 = DFF * D, n2 = D * DFF;
    f32_to_bf16_kernel<<<(nq + 255) / 256, 256, 0, stream>>>(
        wqkv + (size_t)l * nq, wqb, nq);
    f32_to_bf16_kernel<<<(n1 + 255) / 256, 256, 0, stream>>>(
        ffw1 + (size_t)l * n1, w1b, n1);
    f32_to_bf16_kernel<<<(n2 + 255) / 256, 256, 0, stream>>>(
        ffw2 + (size_t)l * n2, w2b, n2);

    // LN1 -> bf16 activations
    ln_kernel<<<ROWS, 256, 0, stream>>>(h, ln1g + l * D, ln1b + l * D, act,
                                        (float*)nullptr, D);
    // QKV projection (q scaled by 1/8 in epilogue)
    dim3 gq(ROWS / GBM, (3 * D) / GBN);
    gemm_bf16_wmma<<<gq, 256, 0, stream>>>(act, wqb, bqkv + (size_t)l * 3 * D,
                                           (float*)nullptr, qkvb, ROWS, 3 * D, D,
                                           /*ep=*/0, /*qcols=*/D);
    // Flash attention; adds into h (residual)
    dim3 ga(T / 64, H, B);
    flash_attn_wmma<<<ga, 256, 0, stream>>>(qkvb, h, T, D);

    // LN2: in-place f32 + bf16 copy for FFN input
    ln_kernel<<<ROWS, 256, 0, stream>>>(h, ln2g + l * D, ln2b + l * D, act, h, D);

    // FFN up + GELU
    dim3 g1(ROWS / GBM, DFF / GBN);
    gemm_bf16_wmma<<<g1, 256, 0, stream>>>(act, w1b, ffb1 + (size_t)l * DFF,
                                           (float*)nullptr, gel, ROWS, DFF, D,
                                           /*ep=*/1, 0);
    // FFN down + residual add into h
    dim3 g2(ROWS / GBM, D / GBN);
    gemm_bf16_wmma<<<g2, 256, 0, stream>>>(gel, w2b, ffb2 + (size_t)l * D, h,
                                           (unsigned short*)nullptr, ROWS, D, DFF,
                                           /*ep=*/2, 0);
  }

  logits_kernel<<<(ROWS + 255) / 256, 256, 0, stream>>>(h, outw, outbb,
                                                        (float*)d_out, ROWS, D);
}
